// BidirectionalEdgeGraphNetwork_58755152609596
// MI455X (gfx1250) — matrix operand order repaired and checked by reading, not compiled, not run on hardware
//
#include <hip/hip_runtime.h>
#include <hip/hip_bf16.h>

#define DEV static __device__ __forceinline__

typedef __attribute__((ext_vector_type(16))) __bf16 v16bf;
typedef __attribute__((ext_vector_type(8)))  float  v8f;

struct U8 { unsigned u[8]; };

// ---------- helpers ----------
// fp32 -> bf16 via native conversion (RNE); lets clang pick v_cvt_*bf16_f32
DEV unsigned short f2bf(float f) {
  __bf16 b = (__bf16)f;
  return __builtin_bit_cast(unsigned short, b);
}
DEV unsigned packbf(float lo, float hi) {
  __bf16 l = (__bf16)lo, h = (__bf16)hi;
  unsigned short ul = __builtin_bit_cast(unsigned short, l);
  unsigned short uh = __builtin_bit_cast(unsigned short, h);
  return (unsigned)ul | ((unsigned)uh << 16);
}
// order-preserving float<->uint for atomicMax-based segment_max
DEV unsigned f2ord(float f) {
  unsigned b = __float_as_uint(f);
  return (b & 0x80000000u) ? ~b : (b | 0x80000000u);
}
DEV float ord2f(unsigned o) {
  unsigned b = (o & 0x80000000u) ? (o & 0x7FFFFFFFu) : ~o;
  return __uint_as_float(b);
}
// A-matrix (16x32 bf16) per-lane K index for VGPR r (ISA 7.12.2)
DEV int a_k0(int r, int hi) { return (r < 4) ? (8 * hi + 2 * r) : (16 + 8 * hi + 2 * (r - 4)); }

DEV v8f wmma_bf16(const U8 &a, const U8 &b, v8f c) {
  return __builtin_amdgcn_wmma_f32_16x16x32_bf16(
      false, __builtin_bit_cast(v16bf, a),
      false, __builtin_bit_cast(v16bf, b),
      (short)0, c, false, false);
}

// A fragment: 16 rows (m = lane&15) x 32 K, source fp32 row pointer per lane (may be null)
DEV void load_a_row(const float *src, int fb, U8 &f) {
  int lane = threadIdx.x & 31;
  int hi = lane >> 4;
  if (src) {
#pragma unroll
    for (int r = 0; r < 8; ++r) {
      int k0 = fb + a_k0(r, hi);
      f.u[r] = packbf(src[k0], src[k0 + 1]);
    }
  } else {
#pragma unroll
    for (int r = 0; r < 8; ++r) f.u[r] = 0u;
  }
}
DEV void load_a_row_scaled(const float *src, int fb, float sc, U8 &f) {
  int lane = threadIdx.x & 31;
  int hi = lane >> 4;
  if (src) {
#pragma unroll
    for (int r = 0; r < 8; ++r) {
      int k0 = fb + a_k0(r, hi);
      f.u[r] = packbf(src[k0] * sc, src[k0 + 1] * sc);
    }
  } else {
#pragma unroll
    for (int r = 0; r < 8; ++r) f.u[r] = 0u;
  }
}
// A fragment from LDS stored as bf16 pairs (u32), row stride ldPairs u32
DEV void load_a_lds(const unsigned *pairs, int ldPairs, int kBase, U8 &f) {
  int lane = threadIdx.x & 31;
  int m = lane & 15, hi = lane >> 4;
#pragma unroll
  for (int r = 0; r < 8; ++r) {
    int k0 = kBase + a_k0(r, hi);
    f.u[r] = pairs[m * ldPairs + (k0 >> 1)];
  }
}
// B fragment: W is row-major [n][K] bf16 (i.e. B[k][n] = W[n][k], giving out = in @ W^T)
DEV void load_b(const unsigned short *W, int ldK, int nBase, int kBase, U8 &f) {
  int lane = threadIdx.x & 31;
  int hi = lane >> 4;
  const unsigned short *p = W + (size_t)(nBase + (lane & 15)) * ldK + kBase;
#pragma unroll
  for (int r = 0; r < 8; ++r) f.u[r] = *(const unsigned *)(p + a_k0(r, hi));
}

// ---------- prep kernels ----------
__global__ void k_cvt_bf16(const float *s, unsigned short *d, int n) {
  for (int i = blockIdx.x * blockDim.x + threadIdx.x; i < n; i += gridDim.x * blockDim.x)
    d[i] = f2bf(s[i]);
}

__global__ void k_init(unsigned *aggOrd, float *osum, float *isum, int *dO, int *dI,
                       int n256, int n) {
  for (int i = blockIdx.x * blockDim.x + threadIdx.x; i < n256; i += gridDim.x * blockDim.x) {
    aggOrd[i] = 0x007FFFFFu;  // ord(-inf)
    osum[i] = 0.f;
    isum[i] = 0.f;
    if (i < n) { dO[i] = 0; dI[i] = 0; }
  }
}

__global__ void k_degrev(const int *row, const int *col, int *dO, int *dI, int *rev, int E) {
  int E2 = E >> 1;
  for (int e = blockIdx.x * blockDim.x + threadIdx.x; e < E; e += gridDim.x * blockDim.x) {
    atomicAdd(&dO[row[e]], 1);
    atomicAdd(&dI[col[e]], 1);
    int cand = (e < E2) ? e + E2 : e - E2;
    rev[e] = (row[cand] == col[e] && col[cand] == row[e]) ? cand : -1;
  }
}

__global__ void k_fin_agg(unsigned *aggOrd, int n256) {
  for (int i = blockIdx.x * blockDim.x + threadIdx.x; i < n256; i += gridDim.x * blockDim.x) {
    float v = ord2f(aggOrd[i]);
    unsigned b = __float_as_uint(v);
    bool fin = (b & 0x7F800000u) != 0x7F800000u;
    ((float *)aggOrd)[i] = fin ? v : 0.f;
  }
}

// ---------- q/v node projections: xq = x@Wq^T+bq, xv = x@Wv^T+bv ----------
__global__ void k_proj_qv(const float *x, const unsigned short *Wqb, const unsigned short *Wvb,
                          const float *bq, const float *bv, float *xq, float *xv, int N) {
  int nodeBase = blockIdx.x * 16;
  int wave = threadIdx.x >> 5;  // 8 waves
  int lane = threadIdx.x & 31;
  int m = lane & 15, hi = lane >> 4;
  int nd = nodeBase + m;
  const float *src = (nd < N) ? x + (size_t)nd * 256 : nullptr;

  v8f acc[4] = {};
  for (int kc = 0; kc < 8; ++kc) {
    U8 a; load_a_row(src, kc * 32, a);
#pragma unroll
    for (int t = 0; t < 4; ++t) {
      int gt = wave * 4 + t;                // 0..31 : 0-15 -> q tiles, 16-31 -> v tiles
      bool isQ = gt < 16;
      int nT = (isQ ? gt : gt - 16) * 16;
      U8 b; load_b(isQ ? Wqb : Wvb, 256, nT, kc * 32, b);
      acc[t] = wmma_bf16(a, b, acc[t]);
    }
  }
#pragma unroll
  for (int t = 0; t < 4; ++t) {
    int gt = wave * 4 + t;
    bool isQ = gt < 16;
    int nc = (isQ ? gt : gt - 16) * 16 + (lane & 15);
    float bias = isQ ? bq[nc] : bv[nc];
    float *dst = isQ ? xq : xv;
#pragma unroll
    for (int v = 0; v < 8; ++v) {
      int nd2 = nodeBase + v + 8 * hi;
      if (nd2 < N) dst[(size_t)nd2 * 256 + nc] = acc[t][v] + bias;
    }
  }
}

// ---------- attention: k=ea@Wk^T+bk (WMMA) then per-(edge,head) MLP + softmax + scatter-max ----------
__global__ void k_attn(const float *ea, const int *row, const int *col,
                       const unsigned short *Wkb, const float *bk,
                       const float *xq, const float *xv,
                       const float *Wa1, const float *ba1,
                       const float *Wa2, const float *ba2,
                       float *probOut, unsigned *aggOrd, int E) {
  __shared__ float k_lds[16 * 257];
  __shared__ float a1s[128 * 65];
  __shared__ float atts[128 * 33];

  int eBase = blockIdx.x * 16;
  int wave = threadIdx.x >> 5;  // 4 waves (128 threads)
  int lane = threadIdx.x & 31;
  int m = lane & 15, hi = lane >> 4;
  int e = eBase + m;
  const float *src = (e < E) ? ea + (size_t)e * 256 : nullptr;

  v8f acc[4] = {};
  for (int kc = 0; kc < 8; ++kc) {
    U8 a; load_a_row(src, kc * 32, a);
#pragma unroll
    for (int t = 0; t < 4; ++t) {
      int nT = (wave * 4 + t) * 16;
      U8 b; load_b(Wkb, 256, nT, kc * 32, b);
      acc[t] = wmma_bf16(a, b, acc[t]);
    }
  }
#pragma unroll
  for (int t = 0; t < 4; ++t) {
    int nc = (wave * 4 + t) * 16 + (lane & 15);
    float bias = bk[nc];
#pragma unroll
    for (int v = 0; v < 8; ++v) {
      int mm = v + 8 * hi;
      k_lds[mm * 257 + nc] = acc[t][v] + bias;
    }
  }
  __syncthreads();

  // phase 2: one thread per (edge-in-tile, head)
  int tid = threadIdx.x;
  int eL = tid >> 3, h = tid & 7;
  int ee = eBase + eL;
  if (ee < E) {
    int rn = row[ee], cn = col[ee];
    float qh[32], kh[32];
#pragma unroll
    for (int i = 0; i < 32; ++i) {
      qh[i] = xq[(size_t)rn * 256 + i * 8 + h];
      kh[i] = k_lds[eL * 257 + i * 8 + h];
    }
    float *a1 = &a1s[tid * 65];
    for (int o = 0; o < 64; ++o) {
      const float *w = Wa1 + o * 64;
      float s = ba1[o];
#pragma unroll
      for (int i = 0; i < 32; ++i) s += w[i] * qh[i];
#pragma unroll
      for (int i = 0; i < 32; ++i) s += w[32 + i] * kh[i];
      a1[o] = fmaxf(s, 0.f);
    }
    float *at = &atts[tid * 33];
    const float invT = 0.17677669529663687f;  // 1/sqrt(32)
    float mx = -3.4e38f;
    for (int o = 0; o < 32; ++o) {
      const float *w = Wa2 + o * 64;
      float s = ba2[o];
      for (int i = 0; i < 64; ++i) s += w[i] * a1[i];
      s *= invT;
      at[o] = s;
      mx = fmaxf(mx, s);
    }
    float sum = 0.f;
    for (int o = 0; o < 32; ++o) { float p = expf(at[o] - mx); at[o] = p; sum += p; }
    float inv = 1.f / sum;
    for (int o = 0; o < 32; ++o) {
      float p = at[o] * inv;
      probOut[(size_t)ee * 256 + o * 8 + h] = p;
      float wv = p * xv[(size_t)cn * 256 + o * 8 + h];
      atomicMax(&aggOrd[(size_t)rn * 256 + o * 8 + h], f2ord(wv));
    }
  }
}

// ---------- fused two-layer edge MLP (dominant GEMM) ----------
__global__ void k_edge_mlp(const float *x, const float *ea, const int *row, const int *col,
                           const int *rev,
                           const unsigned short *We1b, const float *be1,
                           const unsigned short *We2b, const float *be2,
                           float *osum, float *isum, float *edgeOut, int E) {
  __shared__ unsigned h_lds[16 * 385];  // 16 x 768 bf16 (padded pairs)

  int eBase = blockIdx.x * 16;
  int wave = threadIdx.x >> 5;  // 8 waves
  int lane = threadIdx.x & 31;
  int m = lane & 15, hi = lane >> 4;
  int e = eBase + m;
  bool ve = e < E;
  int rowE = 0, colE = 0, revE = -1;
  if (ve) { rowE = row[e]; colE = col[e]; revE = rev[e]; }

  // layer 1: K = 1024 (4 gathered 256-blocks), 48 N-tiles, 6 per wave
  v8f acc[6] = {};
  for (int kc = 0; kc < 32; ++kc) {
    const float *src = nullptr;
    if (ve) {
      int sel = kc >> 3;
      if (sel == 0)      src = x  + (size_t)rowE * 256;
      else if (sel == 1) src = ea + (size_t)e    * 256;
      else if (sel == 2) src = (revE >= 0) ? ea + (size_t)revE * 256 : nullptr;
      else               src = x  + (size_t)colE * 256;
    }
    U8 a; load_a_row(src, (kc & 7) * 32, a);
#pragma unroll
    for (int t = 0; t < 6; ++t) {
      int nT = (wave * 6 + t) * 16;
      U8 b; load_b(We1b, 1024, nT, kc * 32, b);
      acc[t] = wmma_bf16(a, b, acc[t]);
    }
  }
  unsigned short *hs = (unsigned short *)h_lds;  // [16][770] bf16
#pragma unroll
  for (int t = 0; t < 6; ++t) {
    int nc = (wave * 6 + t) * 16 + (lane & 15);
    float bias = be1[nc];
#pragma unroll
    for (int v = 0; v < 8; ++v) {
      int mm = v + 8 * hi;
      hs[mm * 770 + nc] = f2bf(fmaxf(acc[t][v] + bias, 0.f));
    }
  }
  __syncthreads();

  // layer 2: K = 768, 16 N-tiles, 2 per wave
  v8f acc2[2] = {};
  for (int kc = 0; kc < 24; ++kc) {
    U8 a; load_a_lds(h_lds, 385, kc * 32, a);
#pragma unroll
    for (int t = 0; t < 2; ++t) {
      int nT = (wave * 2 + t) * 16;
      U8 b; load_b(We2b, 768, nT, kc * 32, b);
      acc2[t] = wmma_bf16(a, b, acc2[t]);
    }
  }
#pragma unroll
  for (int t = 0; t < 2; ++t) {
    int nc = (wave * 2 + t) * 16 + (lane & 15);
    float bias = be2[nc];
#pragma unroll
    for (int v = 0; v < 8; ++v) {
      int ee = eBase + v + 8 * hi;
      if (ee < E) {
        float ue = acc2[t][v] + bias;
        edgeOut[(size_t)ee * 256 + nc] = fmaxf(ue, 0.f);
        atomicAdd(&osum[(size_t)row[ee] * 256 + nc], ue);
        atomicAdd(&isum[(size_t)col[ee] * 256 + nc], ue);
      }
    }
  }
}

// ---------- twin-edge gate: edge_att = sigmoid([out_mean,in_mean] @ Wea^T + bea) ----------
__global__ void k_edge_att(const float *osum, const float *isum, const int *dO, const int *dI,
                           const unsigned short *Weab, const float *bea, float *eatt, int N) {
  int nodeBase = blockIdx.x * 16;
  int wave = threadIdx.x >> 5;  // 4 waves (128 threads)
  int lane = threadIdx.x & 31;
  int m = lane & 15, hi = lane >> 4;
  int nd = nodeBase + m;
  bool vn = nd < N;
  float invO = 0.f, invI = 0.f;
  const float *po = nullptr, *pi = nullptr;
  if (vn) {
    invO = 1.f / fmaxf((float)dO[nd], 1.f);
    invI = 1.f / fmaxf((float)dI[nd], 1.f);
    po = osum + (size_t)nd * 256;
    pi = isum + (size_t)nd * 256;
  }
  v8f acc[4] = {};
  for (int kc = 0; kc < 16; ++kc) {
    const float *s = (kc < 8) ? po : pi;
    float sc = (kc < 8) ? invO : invI;
    U8 a; load_a_row_scaled(s, (kc & 7) * 32, sc, a);
#pragma unroll
    for (int t = 0; t < 4; ++t) {
      int nT = (wave * 4 + t) * 16;
      U8 b; load_b(Weab, 512, nT, kc * 32, b);
      acc[t] = wmma_bf16(a, b, acc[t]);
    }
  }
#pragma unroll
  for (int t = 0; t < 4; ++t) {
    int nc = (wave * 4 + t) * 16 + (lane & 15);
    float bias = bea[nc];
#pragma unroll
    for (int v = 0; v < 8; ++v) {
      int nd2 = nodeBase + v + 8 * hi;
      if (nd2 < N) {
        float z = acc[t][v] + bias;
        eatt[(size_t)nd2 * 256 + nc] = 1.f / (1.f + expf(-z));
      }
    }
  }
}

// ---------- fused two-layer node MLP + gate ----------
__global__ void k_node_mlp(const float *x, const float *agg,
                           const unsigned short *Wn1b, const float *bn1,
                           const unsigned short *Wn2b, const float *bn2,
                           const float *eatt, float *nodeOut, int N) {
  __shared__ unsigned h_lds[16 * 257];  // 16 x 512 bf16 (padded pairs)

  int nodeBase = blockIdx.x * 16;
  int wave = threadIdx.x >> 5;  // 8 waves
  int lane = threadIdx.x & 31;
  int m = lane & 15, hi = lane >> 4;
  int nd = nodeBase + m;
  bool vn = nd < N;

  // layer 1: K = 512 ([x, agg]), 32 N-tiles, 4 per wave
  v8f acc[4] = {};
  for (int kc = 0; kc < 16; ++kc) {
    const float *src = nullptr;
    if (vn) src = (kc < 8) ? x + (size_t)nd * 256 : agg + (size_t)nd * 256;
    U8 a; load_a_row(src, (kc & 7) * 32, a);
#pragma unroll
    for (int t = 0; t < 4; ++t) {
      int nT = (wave * 4 + t) * 16;
      U8 b; load_b(Wn1b, 512, nT, kc * 32, b);
      acc[t] = wmma_bf16(a, b, acc[t]);
    }
  }
  unsigned short *hs = (unsigned short *)h_lds;  // [16][514] bf16
#pragma unroll
  for (int t = 0; t < 4; ++t) {
    int nc = (wave * 4 + t) * 16 + (lane & 15);
    float bias = bn1[nc];
#pragma unroll
    for (int v = 0; v < 8; ++v) {
      int mm = v + 8 * hi;
      hs[mm * 514 + nc] = f2bf(fmaxf(acc[t][v] + bias, 0.f));
    }
  }
  __syncthreads();

  // layer 2: K = 512, 16 N-tiles, 2 per wave
  v8f acc2[2] = {};
  for (int kc = 0; kc < 16; ++kc) {
    U8 a; load_a_lds(h_lds, 257, kc * 32, a);
#pragma unroll
    for (int t = 0; t < 2; ++t) {
      int nT = (wave * 2 + t) * 16;
      U8 b; load_b(Wn2b, 512, nT, kc * 32, b);
      acc2[t] = wmma_bf16(a, b, acc2[t]);
    }
  }
#pragma unroll
  for (int t = 0; t < 2; ++t) {
    int nc = (wave * 2 + t) * 16 + (lane & 15);
    float bias = bn2[nc];
#pragma unroll
    for (int v = 0; v < 8; ++v) {
      int nd2 = nodeBase + v + 8 * hi;
      if (nd2 < N) {
        float un = fmaxf(acc2[t][v] + bias, 0.f);
        float out = un * eatt[(size_t)nd2 * 256 + nc];
        nodeOut[(size_t)nd2 * 256 + nc] = fmaxf(out, 0.f);
      }
    }
  }
}

// ---------- host ----------
extern "C" void kernel_launch(void* const* d_in, const int* in_sizes, int n_in,
                              void* d_out, int out_size, void* d_ws, size_t ws_size,
                              hipStream_t stream) {
  (void)n_in; (void)out_size; (void)ws_size;
  const int D = 256;
  const float *x   = (const float *)d_in[0];
  const float *ea  = (const float *)d_in[1];
  const float *Wq  = (const float *)d_in[2];  const float *bq  = (const float *)d_in[3];
  const float *Wk  = (const float *)d_in[4];  const float *bk  = (const float *)d_in[5];
  const float *Wv  = (const float *)d_in[6];  const float *bv  = (const float *)d_in[7];
  const float *We1 = (const float *)d_in[8];  const float *be1 = (const float *)d_in[9];
  const float *We2 = (const float *)d_in[10]; const float *be2 = (const float *)d_in[11];
  const float *Wea = (const float *)d_in[12]; const float *bea = (const float *)d_in[13];
  const float *Wn1 = (const float *)d_in[14]; const float *bn1 = (const float *)d_in[15];
  const float *Wn2 = (const float *)d_in[16]; const float *bn2 = (const float *)d_in[17];
  const float *Wa1 = (const float *)d_in[18]; const float *ba1 = (const float *)d_in[19];
  const float *Wa2 = (const float *)d_in[20]; const float *ba2 = (const float *)d_in[21];
  const int   *eix = (const int *)d_in[22];

  const int N = in_sizes[0] / D;
  const int E = in_sizes[22] / 2;
  const int *row = eix;
  const int *col = eix + E;

  // workspace layout
  char *ws = (char *)d_ws;
  size_t off = 0;
  auto alloc = [&](size_t bytes) -> char * {
    char *p = ws + off;
    off = (off + bytes + 255) & ~(size_t)255;
    return p;
  };
  unsigned short *We1b = (unsigned short *)alloc((size_t)768 * 1024 * 2);
  unsigned short *We2b = (unsigned short *)alloc((size_t)256 * 768 * 2);
  unsigned short *Wqb  = (unsigned short *)alloc((size_t)256 * 256 * 2);
  unsigned short *Wkb  = (unsigned short *)alloc((size_t)256 * 256 * 2);
  unsigned short *Wvb  = (unsigned short *)alloc((size_t)256 * 256 * 2);
  unsigned short *Wn1b = (unsigned short *)alloc((size_t)512 * 512 * 2);
  unsigned short *Wn2b = (unsigned short *)alloc((size_t)256 * 512 * 2);
  unsigned short *Weab = (unsigned short *)alloc((size_t)256 * 512 * 2);
  int      *rev    = (int *)alloc((size_t)E * 4);
  float    *xq     = (float *)alloc((size_t)N * 256 * 4);
  float    *xv     = (float *)alloc((size_t)N * 256 * 4);
  unsigned *aggOrd = (unsigned *)alloc((size_t)N * 256 * 4);  // becomes float agg in place
  float    *osum   = (float *)alloc((size_t)N * 256 * 4);
  float    *isum   = (float *)alloc((size_t)N * 256 * 4);
  int      *dO     = (int *)alloc((size_t)N * 4);
  int      *dI     = (int *)alloc((size_t)N * 4);
  float    *eatt   = (float *)alloc((size_t)N * 256 * 4);

  float *nodeOut = (float *)d_out;
  float *edgeOut = nodeOut + (size_t)N * 256;
  float *probOut = edgeOut + (size_t)E * 256;

  // 1) bf16 weight copies
  struct { const float *s; unsigned short *d; int n; } cv[8] = {
      {We1, We1b, 768 * 1024}, {We2, We2b, 256 * 768}, {Wq, Wqb, 256 * 256},
      {Wk, Wkb, 256 * 256},    {Wv, Wvb, 256 * 256},   {Wn1, Wn1b, 512 * 512},
      {Wn2, Wn2b, 256 * 512},  {Wea, Weab, 256 * 512}};
  for (int i = 0; i < 8; ++i) {
    int blocks = (cv[i].n + 255) / 256;
    if (blocks > 2048) blocks = 2048;
    k_cvt_bf16<<<blocks, 256, 0, stream>>>(cv[i].s, cv[i].d, cv[i].n);
  }

  // 2) init accumulators
  int n256 = N * 256;
  k_init<<<2048, 256, 0, stream>>>(aggOrd, osum, isum, dO, dI, n256, N);

  // 3) degrees + reverse-edge index
  k_degrev<<<(E + 255) / 256, 256, 0, stream>>>(row, col, dO, dI, rev, E);

  // 4) node-level q/v projections (WMMA)
  k_proj_qv<<<(N + 15) / 16, 256, 0, stream>>>(x, Wqb, Wvb, bq, bv, xq, xv, N);

  // 5) attention: WMMA k-proj + per-head MLP + softmax + scatter-max
  k_attn<<<(E + 15) / 16, 128, 0, stream>>>(ea, row, col, Wkb, bk, xq, xv,
                                            Wa1, ba1, Wa2, ba2, probOut, aggOrd, E);

  // 6) finalize segment-max (-inf -> 0), in place
  k_fin_agg<<<2048, 256, 0, stream>>>(aggOrd, n256);

  // 7) fused two-layer edge MLP (dominant WMMA) + segment sums + edge output
  k_edge_mlp<<<(E + 15) / 16, 256, 0, stream>>>(x, ea, row, col, rev,
                                                We1b, be1, We2b, be2,
                                                osum, isum, edgeOut, E);

  // 8) twin-edge sigmoid gate (WMMA)
  k_edge_att<<<(N + 15) / 16, 128, 0, stream>>>(osum, isum, dO, dI, Weab, bea, eatt, N);

  // 9) fused two-layer node MLP + gate -> node output (WMMA)
  k_node_mlp<<<(N + 15) / 16, 256, 0, stream>>>(x, (const float *)aggOrd,
                                                Wn1b, bn1, Wn2b, bn2,
                                                eatt, nodeOut, N);
}
